// MOMENTCLSHead_50027779064265
// MI455X (gfx1250) — matrix operand order repaired and checked by reading, not compiled, not run on hardware
//
#include <hip/hip_runtime.h>
#include <hip/hip_bf16.h>
#include <math.h>

// ---------------- problem constants ----------------
#define B_    128
#define D_    18
#define P_    64
#define F_    1024
#define H_    16
#define HD_   64
#define NCLS  23
#define MHID  4096
#define EPSV  1e-5f
#define NGRP  (B_*P_)      // 8192

typedef __bf16 bf16;
typedef __attribute__((ext_vector_type(16))) __bf16 v16bf;
typedef __attribute__((ext_vector_type(8)))  __bf16 v8bf;
typedef __attribute__((ext_vector_type(4)))  __bf16 v4bf;
typedef __attribute__((ext_vector_type(8)))  float  v8f;

typedef unsigned int u32x4 __attribute__((ext_vector_type(4)));
typedef int          i32x4 __attribute__((ext_vector_type(4)));
typedef int          i32x8 __attribute__((ext_vector_type(8)));

// Tensor Data Mover availability (arity differs across toolchains, see doc)
#if defined(__has_builtin)
#  if __has_builtin(__builtin_amdgcn_tensor_load_to_lds)
#    define HAVE_TDM 1
#  endif
#endif
#ifndef HAVE_TDM
#  define HAVE_TDM 0
#endif

// ---------------------------------------------------------------------------
// Generic row LayerNorm over F_=1024 columns. One block (256 thr) per row.
// ---------------------------------------------------------------------------
__global__ __launch_bounds__(256)
void row_ln_kernel(const float* __restrict__ src, float* __restrict__ dstF,
                   bf16* __restrict__ dstB, const float* __restrict__ g,
                   const float* __restrict__ bb)
{
    __shared__ float red[16];
    const int row = blockIdx.x;
    const int t = threadIdx.x;
    const int wave = t >> 5, lane = t & 31;
    const float4 v = *reinterpret_cast<const float4*>(src + (size_t)row * F_ + t * 4);
    float s  = v.x + v.y + v.z + v.w;
    float sq = v.x*v.x + v.y*v.y + v.z*v.z + v.w*v.w;
    #pragma unroll
    for (int off = 16; off >= 1; off >>= 1) {
        s  += __shfl_down(s,  off, 32);
        sq += __shfl_down(sq, off, 32);
    }
    if (lane == 0) { red[wave] = s; red[8 + wave] = sq; }
    __syncthreads();
    float ts = 0.f, tq = 0.f;
    #pragma unroll
    for (int i = 0; i < 8; ++i) { ts += red[i]; tq += red[8 + i]; }
    const float mu = ts * (1.0f / F_);
    const float rs = rsqrtf(tq * (1.0f / F_) - mu * mu + EPSV);
    const int c = t * 4;
    const float4 gv = *reinterpret_cast<const float4*>(g + c);
    const float4 bv = *reinterpret_cast<const float4*>(bb + c);
    float o0 = (v.x - mu) * rs * gv.x + bv.x;
    float o1 = (v.y - mu) * rs * gv.y + bv.y;
    float o2 = (v.z - mu) * rs * gv.z + bv.z;
    float o3 = (v.w - mu) * rs * gv.w + bv.w;
    if (dstF) {
        float4 o; o.x = o0; o.y = o1; o.z = o2; o.w = o3;
        *reinterpret_cast<float4*>(dstF + (size_t)row * F_ + c) = o;
    }
    if (dstB) {
        v4bf o; o[0] = (bf16)o0; o[1] = (bf16)o1; o[2] = (bf16)o2; o[3] = (bf16)o3;
        *reinterpret_cast<v4bf*>(dstB + (size_t)row * F_ + c) = o;
    }
}

// ---------------------------------------------------------------------------
// qh[c]  = sum_f qln[f]  * ch_in_w[f][c] + ch_in_b[c]   (idx <  1024)
// qh2[c] = sum_f cqln[f] * t_in_w [f][c] + t_in_b [c]   (idx >= 1024)
// ---------------------------------------------------------------------------
__global__ void qh_kernel(const float* __restrict__ qln, const float* __restrict__ cqln,
                          const float* __restrict__ ch_in_w, const float* __restrict__ ch_in_b,
                          const float* __restrict__ t_in_w,  const float* __restrict__ t_in_b,
                          float* __restrict__ qh, float* __restrict__ qh2)
{
    const int idx = blockIdx.x * 256 + threadIdx.x;    // 0..2047
    const int c = idx & (F_ - 1);
    const float* q  = (idx < F_) ? qln     : cqln;
    const float* W  = (idx < F_) ? ch_in_w : t_in_w;
    const float* bb = (idx < F_) ? ch_in_b : t_in_b;
    float s = bb[c];
    for (int f = 0; f < F_; ++f)
        s = fmaf(q[f], W[(size_t)f * 3072 + c], s);
    ((idx < F_) ? qh : qh2)[c] = s;
}

// ---------------------------------------------------------------------------
// Effective query matrices (scale 1/sqrt(HD) folded in).
// ---------------------------------------------------------------------------
__global__ void wqeff_kernel(const float* __restrict__ qh, const float* __restrict__ qh2,
                             const float* __restrict__ ch_in_w, const float* __restrict__ ch_in_b,
                             const float* __restrict__ t_in_w,  const float* __restrict__ t_in_b,
                             float* __restrict__ wq_eff, float* __restrict__ wq2_eff,
                             float* __restrict__ bias_eff, float* __restrict__ bias2_eff)
{
    const int idx = blockIdx.x * 256 + threadIdx.x;
    const float scale = 0.125f;                      // 1/sqrt(64)
    if (idx < 2 * H_ * F_) {
        const int which = idx >> 14;
        const int r = idx & 16383;
        const int h = r >> 10, f = r & 1023;
        const float* q = which ? qh2 : qh;
        const float* W = which ? t_in_w : ch_in_w;
        float s = 0.f;
        #pragma unroll 8
        for (int j = 0; j < HD_; ++j)
            s = fmaf(q[h * HD_ + j], W[(size_t)f * 3072 + F_ + h * HD_ + j], s);
        (which ? wq2_eff : wq_eff)[h * F_ + f] = s * scale;
    } else if (idx < 2 * H_ * F_ + 32) {
        const int r = idx - 2 * H_ * F_;
        const int which = r >> 4, h = r & 15;
        const float* q  = which ? qh2 : qh;
        const float* bb = which ? t_in_b : ch_in_b;
        float s = 0.f;
        for (int j = 0; j < HD_; ++j) s += q[h * HD_ + j] * bb[F_ + h * HD_ + j];
        (which ? bias2_eff : bias_eff)[h] = s * scale;
    }
}

// ---------------------------------------------------------------------------
// fp32 -> bf16 *transposed* slice conversion:  dst[n][k] = src[k][colOff + n]
// (weights become L2-resident N-major bf16, so GEMM B tiles are row-major)
// ---------------------------------------------------------------------------
__global__ void cvt_bf16_t_kernel(const float* __restrict__ src, bf16* __restrict__ dst,
                                  int N, int K, int srcStride, int colOff)
{
    const size_t total = (size_t)N * K;
    for (size_t i = blockIdx.x * (size_t)blockDim.x + threadIdx.x; i < total;
         i += (size_t)gridDim.x * blockDim.x) {
        const size_t n = i / K, k = i - n * K;
        dst[i] = (bf16)src[k * (size_t)srcStride + colOff + n];
    }
}

// ---------------------------------------------------------------------------
// Stage 1: per-(b,p) cross-channel attention. One block per group g=b*P+p.
// ---------------------------------------------------------------------------
__global__ __launch_bounds__(256)
void stage1_kernel(const float* __restrict__ x, const float* __restrict__ wq_eff,
                   const float* __restrict__ bias_eff, bf16* __restrict__ Tbar)
{
    __shared__ __align__(16) bf16 tokLds[D_ * F_];   // 36 KB
    __shared__ float attnLds[D_ * H_];               // scores -> attn
    __shared__ float redLds[8 * H_];

    const int g = blockIdx.x;
    const int b = g >> 6, p = g & 63;
    const int t = threadIdx.x;
    const int wave = t >> 5, lane = t & 31;
    const int c0 = t * 4;

    float4 wq[H_];
    #pragma unroll
    for (int h = 0; h < H_; ++h)
        wq[h] = *reinterpret_cast<const float4*>(wq_eff + h * F_ + c0);

    for (int d = 0; d < D_; ++d) {
        const float4 tv = *reinterpret_cast<const float4*>(
            x + (((size_t)b * D_ + d) * P_ + p) * F_ + c0);
        if (d + 1 < D_)   // emits global_prefetch_b8 on gfx1250
            __builtin_prefetch(x + (((size_t)b * D_ + d + 1) * P_ + p) * F_ + c0, 0, 1);
        v4bf tkb; tkb[0] = (bf16)tv.x; tkb[1] = (bf16)tv.y;
        tkb[2] = (bf16)tv.z; tkb[3] = (bf16)tv.w;
        *reinterpret_cast<v4bf*>(tokLds + d * F_ + c0) = tkb;

        #pragma unroll
        for (int h = 0; h < H_; ++h) {
            float v = fmaf(tv.x, wq[h].x, fmaf(tv.y, wq[h].y,
                      fmaf(tv.z, wq[h].z, tv.w * wq[h].w)));
            v += __shfl_down(v, 16, 32); v += __shfl_down(v, 8, 32);
            v += __shfl_down(v, 4, 32);  v += __shfl_down(v, 2, 32);
            v += __shfl_down(v, 1, 32);
            if (lane == 0) redLds[wave * H_ + h] = v;
        }
        __syncthreads();
        if (t < H_) {
            float s = bias_eff[t];
            #pragma unroll
            for (int w = 0; w < 8; ++w) s += redLds[w * H_ + t];
            attnLds[d * H_ + t] = s;
        }
        __syncthreads();
    }

    if (t < H_) {   // softmax over d (18) per head
        float m = -3.4e38f;
        for (int d = 0; d < D_; ++d) m = fmaxf(m, attnLds[d * H_ + t]);
        float s = 0.f;
        for (int d = 0; d < D_; ++d) {
            float e = __expf(attnLds[d * H_ + t] - m);
            attnLds[d * H_ + t] = e; s += e;
        }
        const float inv = 1.0f / s;
        for (int d = 0; d < D_; ++d) attnLds[d * H_ + t] *= inv;
    }
    __syncthreads();

    float tb[H_][4];
    #pragma unroll
    for (int h = 0; h < H_; ++h)
        { tb[h][0] = 0.f; tb[h][1] = 0.f; tb[h][2] = 0.f; tb[h][3] = 0.f; }
    for (int d = 0; d < D_; ++d) {
        float tk[4];
        #pragma unroll
        for (int j = 0; j < 4; ++j) tk[j] = (float)tokLds[d * F_ + c0 + j];
        #pragma unroll
        for (int h = 0; h < H_; ++h) {
            const float a = attnLds[d * H_ + h];
            #pragma unroll
            for (int j = 0; j < 4; ++j) tb[h][j] = fmaf(a, tk[j], tb[h][j]);
        }
    }
    #pragma unroll
    for (int h = 0; h < H_; ++h) {
        v4bf o;
        #pragma unroll
        for (int j = 0; j < 4; ++j) o[j] = (bf16)tb[h][j];
        *reinterpret_cast<v4bf*>(Tbar + ((size_t)h * NGRP + g) * F_ + c0) = o;
    }
}

// ---------------------------------------------------------------------------
// TDM: issue a 2D tensor_load_to_lds of an nRows x 32 bf16 tile (row-major,
// row stride = rowStride elements). pad_enable adds 4 DWORDs (16B) after
// every 16 DWORDs (64B = one 32-elem row) => LDS row stride 80B = 40
// halfwords, exactly the conflict-free layout the WMMA fragment loads use.
// Issued by one wave; tracked on TENSORcnt.
// ---------------------------------------------------------------------------
#if HAVE_TDM
__device__ __forceinline__ void tdm_load_tile(const bf16* gsrc, unsigned ldsByteAddr,
                                              int rowStride /*elements*/, int nRows)
{
    const unsigned long long ga = (unsigned long long)(size_t)gsrc;
    u32x4 g0;
    g0[0] = 1u;                                        // count=1 (valid user D#)
    g0[1] = ldsByteAddr;                               // lds_addr (bits 63:32)
    g0[2] = (unsigned)(ga & 0xFFFFFFFFull);            // global_addr[31:0]
    g0[3] = (unsigned)((ga >> 32) & 0x1FFFFFFull)      // global_addr[56:32]
          | 0x80000000u;                               // type=2 ("image")
    const unsigned td0 = 1u << 30, td1 = 1u << 30;     // huge dims: no OOB clamp
    const unsigned tile0 = 32u;
    const unsigned tile1 = (unsigned)nRows;
    const unsigned s0 = (unsigned)rowStride;           // dim0 stride (elements)
    i32x8 g1;
    g1[0] = (int)((1u << 16)                           // data_size = 2 bytes
                | (1u << 20)                           // pad_enable
                | (3u << 22)                           // pad_interval = 16 dw
                | (3u << 25));                         // pad_amount   = 4 dw
    g1[1] = (int)((td0 & 0xFFFFu) << 16);              // tensor_dim0[15:0]
    g1[2] = (int)((td0 >> 16) | ((td1 & 0xFFFFu) << 16));
    g1[3] = (int)((td1 >> 16) | (tile0 << 16));        // tensor_dim1 hi | tile0
    g1[4] = (int)tile1;                                // tile_dim1, tile_dim2=0
    g1[5] = (int)s0;                                   // dim0_stride[31:0]
    g1[6] = 0;
    g1[7] = 0;
    i32x4 gz; gz[0] = 0; gz[1] = 0; gz[2] = 0; gz[3] = 0;
#if __clang_major__ >= 23
    i32x8 gz8;
    #pragma unroll
    for (int i = 0; i < 8; ++i) gz8[i] = 0;
    __builtin_amdgcn_tensor_load_to_lds(g0, g1, gz, gz, gz8, 0);
#else
    __builtin_amdgcn_tensor_load_to_lds(g0, g1, gz, gz, 0);
#endif
}
#endif

// ---------------------------------------------------------------------------
// bf16 WMMA GEMM:  C(z) = A(z) @ Bt(z)^T + bias(z)   [+GELU] [bf16|f32 out]
// A is M x K row-major; Bt is N x K row-major (pre-transposed weights).
// Block tile 128(M) x 64(N), K step 32, 8 waves, double-buffered LDS.
// Both tiles staged by the Tensor Data Mover (one wave issues 2 descriptors
// per K-step, everyone else computes); one barrier per K-step.
// ---------------------------------------------------------------------------
template<bool GELU, bool OUTB>
__global__ __launch_bounds__(256)
void gemm_bf16_kernel(const bf16* __restrict__ A, const bf16* __restrict__ Bt,
                      const float* __restrict__ bias, void* __restrict__ Cv,
                      int K, int lda, int ldb, int ldc,
                      long long sAz, long long sBz, long long sCz, long long sBiasz)
{
    constexpr int LDS_S = 40;
    __shared__ __align__(16) bf16 As[2][128 * LDS_S];
    __shared__ __align__(16) bf16 Bs[2][64 * LDS_S];

    A  += (size_t)blockIdx.z * sAz;
    Bt += (size_t)blockIdx.z * sBz;
    const int m0 = blockIdx.x * 128;
    const int n0 = blockIdx.y * 64;

    const int t = threadIdx.x;
    const int wave = t >> 5, lane = t & 31;
    const int half = lane >> 4, r = lane & 15;
    const int arow = t >> 1, aseg = (t & 1) * 16;     // A loader mapping (fallback)
    const int brow = t >> 2, bseg = (t & 3) * 8;      // B loader mapping (fallback)
    (void)arow; (void)aseg; (void)brow; (void)bseg;

    v8f acc[4];
    #pragma unroll
    for (int nt = 0; nt < 4; ++nt)
        #pragma unroll
        for (int i = 0; i < 8; ++i) acc[nt][i] = 0.f;

    const int nIter = K >> 5;

    // ---- prologue: stage tile pair 0 into buffer 0 ----
#if HAVE_TDM
    if (wave == 0) {
        tdm_load_tile(A  + (size_t)m0 * lda, (unsigned)(size_t)(&As[0][0]), lda, 128);
        tdm_load_tile(Bt + (size_t)n0 * ldb, (unsigned)(size_t)(&Bs[0][0]), ldb, 64);
        __builtin_amdgcn_s_wait_tensorcnt((short)0);
    }
#else
    {
        const bf16* srcA = A + (size_t)(m0 + arow) * lda + aseg;
        *reinterpret_cast<v8bf*>(&As[0][arow * LDS_S + aseg]) =
            *reinterpret_cast<const v8bf*>(srcA);
        *reinterpret_cast<v8bf*>(&As[0][arow * LDS_S + aseg + 8]) =
            *reinterpret_cast<const v8bf*>(srcA + 8);
        const bf16* srcB = Bt + (size_t)(n0 + brow) * ldb + bseg;
        *reinterpret_cast<v8bf*>(&Bs[0][brow * LDS_S + bseg]) =
            *reinterpret_cast<const v8bf*>(srcB);
    }
#endif
    __syncthreads();

    int buf = 0;
    for (int it = 0; it < nIter; ++it) {
        const int nbuf = buf ^ 1;
        const bool hasNext = (it + 1 < nIter);
        const int kn = (it + 1) << 5;
#if HAVE_TDM
        if (hasNext && wave == 0) {
            tdm_load_tile(A + (size_t)m0 * lda + kn,
                          (unsigned)(size_t)(&As[nbuf][0]), lda, 128);
            tdm_load_tile(Bt + (size_t)n0 * ldb + kn,
                          (unsigned)(size_t)(&Bs[nbuf][0]), ldb, 64);
        }
#else
        v8bf an0, an1, bn0;
        if (hasNext) {
            const bf16* srcA = A + (size_t)(m0 + arow) * lda + kn + aseg;
            an0 = *reinterpret_cast<const v8bf*>(srcA);
            an1 = *reinterpret_cast<const v8bf*>(srcA + 8);
            bn0 = *reinterpret_cast<const v8bf*>(
                Bt + (size_t)(n0 + brow) * ldb + kn + bseg);
        }
#endif

        // ---- compute on current buffer: 1 A frag + 4 B frags, 4 WMMA ----
        const bf16* curA = &As[buf][0];
        const bf16* curB = &Bs[buf][0];
        const bf16* ap = curA + (wave * 16 + r) * LDS_S + half * 8;
        const v8bf alo = *reinterpret_cast<const v8bf*>(ap);
        const v8bf ahi = *reinterpret_cast<const v8bf*>(ap + 16);
        v16bf af;
        #pragma unroll
        for (int i = 0; i < 8; ++i) { af[i] = alo[i]; af[i + 8] = ahi[i]; }

        v16bf bfr[4];
        #pragma unroll
        for (int nt = 0; nt < 4; ++nt) {
            const bf16* bp = curB + (nt * 16 + r) * LDS_S + half * 8;
            const v8bf blo = *reinterpret_cast<const v8bf*>(bp);
            const v8bf bhi = *reinterpret_cast<const v8bf*>(bp + 16);
            #pragma unroll
            for (int i = 0; i < 8; ++i) { bfr[nt][i] = blo[i]; bfr[nt][i + 8] = bhi[i]; }
        }
        #pragma unroll
        for (int nt = 0; nt < 4; ++nt)
            acc[nt] = __builtin_amdgcn_wmma_f32_16x16x32_bf16(
                false, af, false, bfr[nt], (short)0, acc[nt], false, false);

        // ---- finish staging next tile pair ----
        if (hasNext) {
#if HAVE_TDM
            if (wave == 0) __builtin_amdgcn_s_wait_tensorcnt((short)0);
#else
            *reinterpret_cast<v8bf*>(&As[nbuf][arow * LDS_S + aseg])     = an0;
            *reinterpret_cast<v8bf*>(&As[nbuf][arow * LDS_S + aseg + 8]) = an1;
            *reinterpret_cast<v8bf*>(&Bs[nbuf][brow * LDS_S + bseg])     = bn0;
#endif
        }
        __syncthreads();
        buf = nbuf;
    }

    float* Cf = (float*)Cv + (OUTB ? 0 : (size_t)blockIdx.z * sCz);
    bf16*  Cb = (bf16*)Cv + (OUTB ? (size_t)blockIdx.z * sCz : 0);
    const float* bp2 = bias ? (bias + (size_t)blockIdx.z * sBiasz) : nullptr;
    #pragma unroll
    for (int nt = 0; nt < 4; ++nt) {
        const int n = n0 + nt * 16 + r;
        const float bv = bp2 ? bp2[n] : 0.f;
        #pragma unroll
        for (int r8 = 0; r8 < 8; ++r8) {
            const int m = m0 + wave * 16 + half * 8 + r8;
            float v = acc[nt][r8] + bv;
            if (GELU) v = 0.5f * v * (1.0f + erff(v * 0.70710678118654752f));
            if (OUTB) Cb[(size_t)m * ldc + n] = (bf16)v;
            else      Cf[(size_t)m * ldc + n] = v;
        }
    }
}

// ---------------------------------------------------------------------------
// Stage 2: temporal attention, one block per batch b.
// ---------------------------------------------------------------------------
__global__ __launch_bounds__(256)
void stage2_kernel(const float* __restrict__ fused, const float* __restrict__ wq2_eff,
                   const float* __restrict__ bias2_eff,
                   const float* __restrict__ t_in_w, const float* __restrict__ t_in_b,
                   const float* __restrict__ pnp_g, const float* __restrict__ pnp_b,
                   bf16* __restrict__ pooled2B)
{
    __shared__ float muL[P_], rsL[P_];
    __shared__ float attn2[P_ * H_];
    __shared__ float redLds[8 * H_];
    __shared__ __align__(16) bf16 t2L[H_ * F_];      // 32 KB

    const int b = blockIdx.x;
    const int t = threadIdx.x;
    const int wave = t >> 5, lane = t & 31;
    const int c0 = t * 4;
    const float* fb = fused + (size_t)b * P_ * F_;

    for (int p = wave; p < P_; p += 8) {
        float s = 0.f, sq = 0.f;
        #pragma unroll
        for (int cc = 0; cc < 8; ++cc) {
            const float4 v = *reinterpret_cast<const float4*>(
                fb + (size_t)p * F_ + lane * 4 + cc * 128);
            s  += v.x + v.y + v.z + v.w;
            sq += v.x*v.x + v.y*v.y + v.z*v.z + v.w*v.w;
        }
        #pragma unroll
        for (int off = 16; off >= 1; off >>= 1) {
            s += __shfl_down(s, off, 32); sq += __shfl_down(sq, off, 32);
        }
        if (lane == 0) {
            const float mu = s * (1.0f / F_);
            muL[p] = mu;
            rsL[p] = rsqrtf(sq * (1.0f / F_) - mu * mu + EPSV);
        }
    }
    __syncthreads();

    float4 wq[H_];
    #pragma unroll
    for (int h = 0; h < H_; ++h)
        wq[h] = *reinterpret_cast<const float4*>(wq2_eff + h * F_ + c0);
    const float4 g4 = *reinterpret_cast<const float4*>(pnp_g + c0);
    const float4 b4 = *reinterpret_cast<const float4*>(pnp_b + c0);

    for (int p = 0; p < P_; ++p) {
        const float4 v = *reinterpret_cast<const float4*>(fb + (size_t)p * F_ + c0);
        const float mu = muL[p], rs = rsL[p];
        const float k0v = (v.x - mu) * rs * g4.x + b4.x;
        const float k1v = (v.y - mu) * rs * g4.y + b4.y;
        const float k2v = (v.z - mu) * rs * g4.z + b4.z;
        const float k3v = (v.w - mu) * rs * g4.w + b4.w;
        #pragma unroll
        for (int h = 0; h < H_; ++h) {
            float s = fmaf(k0v, wq[h].x, fmaf(k1v, wq[h].y,
                      fmaf(k2v, wq[h].z, k3v * wq[h].w)));
            s += __shfl_down(s, 16, 32); s += __shfl_down(s, 8, 32);
            s += __shfl_down(s, 4, 32);  s += __shfl_down(s, 2, 32);
            s += __shfl_down(s, 1, 32);
            if (lane == 0) redLds[wave * H_ + h] = s;
        }
        __syncthreads();
        if (t < H_) {
            float s = bias2_eff[t];
            #pragma unroll
            for (int w = 0; w < 8; ++w) s += redLds[w * H_ + t];
            attn2[p * H_ + t] = s;
        }
        __syncthreads();
    }

    if (t < H_) {   // softmax over P
        float m = -3.4e38f;
        for (int p = 0; p < P_; ++p) m = fmaxf(m, attn2[p * H_ + t]);
        float s = 0.f;
        for (int p = 0; p < P_; ++p) {
            float e = __expf(attn2[p * H_ + t] - m);
            attn2[p * H_ + t] = e; s += e;
        }
        const float inv = 1.0f / s;
        for (int p = 0; p < P_; ++p) attn2[p * H_ + t] *= inv;
    }
    __syncthreads();

    float t2[H_][4];
    #pragma unroll
    for (int h = 0; h < H_; ++h)
        { t2[h][0] = 0.f; t2[h][1] = 0.f; t2[h][2] = 0.f; t2[h][3] = 0.f; }
    for (int p = 0; p < P_; ++p) {
        const float4 v = *reinterpret_cast<const float4*>(fb + (size_t)p * F_ + c0);
        const float mu = muL[p], rs = rsL[p];
        float kv[4];
        kv[0] = (v.x - mu) * rs * g4.x + b4.x;
        kv[1] = (v.y - mu) * rs * g4.y + b4.y;
        kv[2] = (v.z - mu) * rs * g4.z + b4.z;
        kv[3] = (v.w - mu) * rs * g4.w + b4.w;
        #pragma unroll
        for (int h = 0; h < H_; ++h) {
            const float a = attn2[p * H_ + h];
            #pragma unroll
            for (int j = 0; j < 4; ++j) t2[h][j] = fmaf(a, kv[j], t2[h][j]);
        }
    }
    #pragma unroll
    for (int h = 0; h < H_; ++h)
        #pragma unroll
        for (int j = 0; j < 4; ++j) t2L[h * F_ + c0 + j] = (bf16)t2[h][j];
    __syncthreads();

    const int o0 = t * 4;
    const int h = o0 >> 6;
    float accp[4];
    #pragma unroll
    for (int j = 0; j < 4; ++j) accp[j] = t_in_b[2 * F_ + o0 + j];
    for (int c = 0; c < F_; ++c) {
        const float tv = (float)t2L[h * F_ + c];
        const float4 wr = *reinterpret_cast<const float4*>(
            t_in_w + (size_t)c * 3072 + 2 * F_ + o0);
        accp[0] = fmaf(tv, wr.x, accp[0]); accp[1] = fmaf(tv, wr.y, accp[1]);
        accp[2] = fmaf(tv, wr.z, accp[2]); accp[3] = fmaf(tv, wr.w, accp[3]);
    }
    v4bf o;
    #pragma unroll
    for (int j = 0; j < 4; ++j) o[j] = (bf16)accp[j];
    *reinterpret_cast<v4bf*>(pooled2B + (size_t)b * F_ + o0) = o;
}

// ---------------------------------------------------------------------------
// logits[b][cls] = sum_k hact[b][k] * w2[k][cls] + b2[cls]
// ---------------------------------------------------------------------------
__global__ __launch_bounds__(256)
void logits_kernel(const bf16* __restrict__ hact, const float* __restrict__ w2,
                   const float* __restrict__ b2, float* __restrict__ out)
{
    __shared__ float redL[8 * NCLS];
    const int b = blockIdx.x;
    const int t = threadIdx.x;
    const int wave = t >> 5, lane = t & 31;
    float acc[NCLS];
    #pragma unroll
    for (int c = 0; c < NCLS; ++c) acc[c] = 0.f;
    for (int k = t; k < MHID; k += 256) {
        const float hv = (float)hact[(size_t)b * MHID + k];
        const float* wr = w2 + (size_t)k * NCLS;
        #pragma unroll
        for (int c = 0; c < NCLS; ++c) acc[c] = fmaf(hv, wr[c], acc[c]);
    }
    #pragma unroll
    for (int c = 0; c < NCLS; ++c) {
        float v = acc[c];
        v += __shfl_down(v, 16, 32); v += __shfl_down(v, 8, 32);
        v += __shfl_down(v, 4, 32);  v += __shfl_down(v, 2, 32);
        v += __shfl_down(v, 1, 32);
        if (lane == 0) redL[wave * NCLS + c] = v;
    }
    __syncthreads();
    if (t < NCLS) {
        float s = b2[t];
        #pragma unroll
        for (int w = 0; w < 8; ++w) s += redL[w * NCLS + t];
        out[(size_t)b * NCLS + t] = s;
    }
}

// ---------------------------------------------------------------------------
extern "C" void kernel_launch(void* const* d_in, const int* in_sizes, int n_in,
                              void* d_out, int out_size, void* d_ws, size_t ws_size,
                              hipStream_t stream)
{
    (void)in_sizes; (void)n_in; (void)out_size; (void)ws_size;
    const float* x       = (const float*)d_in[0];
    const float* ch_q    = (const float*)d_in[1];
    const float* cls_q   = (const float*)d_in[2];
    const float* ch_in_w = (const float*)d_in[3];
    const float* ch_in_b = (const float*)d_in[4];
    const float* ch_o_w  = (const float*)d_in[5];
    const float* ch_o_b  = (const float*)d_in[6];
    const float* t_in_w  = (const float*)d_in[7];
    const float* t_in_b  = (const float*)d_in[8];
    const float* t_o_w   = (const float*)d_in[9];
    const float* t_o_b   = (const float*)d_in[10];
    const float* qn_g    = (const float*)d_in[11];
    const float* qn_b    = (const float*)d_in[12];
    const float* cn_g    = (const float*)d_in[13];
    const float* cn_b    = (const float*)d_in[14];
    const float* pnc_g   = (const float*)d_in[15];
    const float* pnc_b   = (const float*)d_in[16];
    const float* pnp_g   = (const float*)d_in[17];
    const float* pnp_b   = (const float*)d_in[18];
    const float* pon_g   = (const float*)d_in[19];
    const float* pon_b   = (const float*)d_in[20];
    const float* w1      = (const float*)d_in[21];
    const float* b1      = (const float*)d_in[22];
    const float* w2      = (const float*)d_in[23];
    const float* b2      = (const float*)d_in[24];
    float* out = (float*)d_out;

    char* base = (char*)d_ws;
    size_t off = 0;
    auto take = [&](size_t n) -> void* {
        void* p = base + off;
        off = (off + n + 255) & ~(size_t)255;
        return p;
    };
    float* qln      = (float*)take(F_ * 4);
    float* cqln     = (float*)take(F_ * 4);
    float* qh       = (float*)take(F_ * 4);
    float* qh2      = (float*)take(F_ * 4);
    float* bias_e   = (float*)take(H_ * 4);
    float* bias2_e  = (float*)take(H_ * 4);
    float* wq_eff   = (float*)take((size_t)H_ * F_ * 4);
    float* wq2_eff  = (float*)take((size_t)H_ * F_ * 4);
    bf16*  Wv1t     = (bf16*)take((size_t)F_ * F_ * 2);      // N x K
    bf16*  Woutt    = (bf16*)take((size_t)F_ * F_ * 2);      // N x K
    bf16*  Toutt    = (bf16*)take((size_t)F_ * F_ * 2);      // N x K
    bf16*  W1t      = (bf16*)take((size_t)MHID * F_ * 2);    // N x K
    bf16*  Tbar     = (bf16*)take((size_t)H_ * NGRP * F_ * 2);   // 256 MB
    bf16*  pooledB  = (bf16*)take((size_t)NGRP * F_ * 2);
    float* fusedPre = (float*)take((size_t)NGRP * F_ * 4);
    float* fused    = (float*)take((size_t)NGRP * F_ * 4);
    bf16*  pooled2B = (bf16*)take((size_t)B_ * F_ * 2);
    float* zPre     = (float*)take((size_t)B_ * F_ * 4);
    bf16*  zB       = (bf16*)take((size_t)B_ * F_ * 2);
    bf16*  hact     = (bf16*)take((size_t)B_ * MHID * 2);

    row_ln_kernel<<<1, 256, 0, stream>>>(ch_q,  qln,  nullptr, qn_g,  qn_b);
    row_ln_kernel<<<1, 256, 0, stream>>>(cls_q, cqln, nullptr, pnc_g, pnc_b);
    qh_kernel<<<8, 256, 0, stream>>>(qln, cqln, ch_in_w, ch_in_b, t_in_w, t_in_b, qh, qh2);
    wqeff_kernel<<<129, 256, 0, stream>>>(qh, qh2, ch_in_w, ch_in_b, t_in_w, t_in_b,
                                          wq_eff, wq2_eff, bias_e, bias2_e);
    // weights -> bf16, transposed to N-major (B tiles become row-major)
    cvt_bf16_t_kernel<<<2048, 256, 0, stream>>>(ch_in_w, Wv1t, F_, F_, 3072, 2 * F_);
    cvt_bf16_t_kernel<<<2048, 256, 0, stream>>>(ch_o_w,  Woutt, F_, F_, F_, 0);
    cvt_bf16_t_kernel<<<2048, 256, 0, stream>>>(t_o_w,   Toutt, F_, F_, F_, 0);
    cvt_bf16_t_kernel<<<2048, 256, 0, stream>>>(w1,      W1t,  MHID, F_, MHID, 0);

    stage1_kernel<<<NGRP, 256, 0, stream>>>(x, wq_eff, bias_e, Tbar);

    // G1: per-head V projection; head h uses rows 64h..64h+63 of Wv1t
    gemm_bf16_kernel<false, true><<<dim3(NGRP / 128, 1, H_), 256, 0, stream>>>(
        Tbar, Wv1t, ch_in_b + 2 * F_, pooledB,
        F_, F_, F_, F_,
        (long long)NGRP * F_, (long long)64 * F_, 64, 64);

    gemm_bf16_kernel<false, false><<<dim3(NGRP / 128, F_ / 64, 1), 256, 0, stream>>>(
        pooledB, Woutt, ch_o_b, fusedPre, F_, F_, F_, F_, 0, 0, 0, 0);
    row_ln_kernel<<<NGRP, 256, 0, stream>>>(fusedPre, fused, nullptr, cn_g, cn_b);

    stage2_kernel<<<B_, 256, 0, stream>>>(fused, wq2_eff, bias2_e,
                                          t_in_w, t_in_b, pnp_g, pnp_b, pooled2B);

    gemm_bf16_kernel<false, false><<<dim3(1, F_ / 64, 1), 256, 0, stream>>>(
        pooled2B, Toutt, t_o_b, zPre, F_, F_, F_, F_, 0, 0, 0, 0);
    row_ln_kernel<<<B_, 256, 0, stream>>>(zPre, nullptr, zB, pon_g, pon_b);

    gemm_bf16_kernel<true, true><<<dim3(1, MHID / 64, 1), 256, 0, stream>>>(
        zB, W1t, b1, hact, F_, F_, MHID, MHID, 0, 0, 0, 0);

    logits_kernel<<<B_, 256, 0, stream>>>(hact, w2, b2, out);
}
